// LM_23304492548274
// MI455X (gfx1250) — matrix operand order repaired and checked by reading, not compiled, not run on hardware
//
#include <hip/hip_runtime.h>
#include <hip/hip_bf16.h>

// Problem constants (match reference)
#define BB   8
#define TT   4096
#define MM   (BB * TT)     // 32768 rows
#define DD   384           // model dim
#define NN   1024          // state dim
#define VV   256           // vocab
#define CH   16            // scan chunks per sequence
#define TC   (TT / CH)     // 256 steps per chunk

#define LDS_PITCH 40       // bf16 elems per staged row (80B): conflict-free, 16B-aligned

#if defined(__has_builtin)
#if __has_builtin(__builtin_amdgcn_global_load_async_to_lds_b128)
#define USE_ASYNC_LDS 1
#endif
#endif

typedef __attribute__((ext_vector_type(16))) __bf16 v16bf;
typedef __attribute__((ext_vector_type(8)))  __bf16 v8bf;
typedef __attribute__((ext_vector_type(8)))  float  v8f;
typedef __attribute__((ext_vector_type(4)))  int    v4i;

union Frag { v16bf f; v8bf h[2]; };

#if USE_ASYNC_LDS
typedef __attribute__((address_space(1))) v4i* as1_v4i;
typedef __attribute__((address_space(3))) v4i* as3_v4i;
// 16B async DMA global -> LDS (ASYNCcnt-tracked, bypasses VGPRs).
// AS3 pointer = low 32 bits of the flat LDS address (aperture lives in the
// high half per the ISA flat-address layout).
__device__ __forceinline__ void async_cp16(const void* g, void* l) {
    __builtin_amdgcn_global_load_async_to_lds_b128(
        (as1_v4i)(uintptr_t)g, (as3_v4i)(unsigned)(uintptr_t)l, 0, 0);
}
#endif

// ---------------------------------------------------------------------------
// f32 -> bf16 flat convert (weights)
// ---------------------------------------------------------------------------
__global__ void cvt_f32_bf16(const float* __restrict__ src,
                             __hip_bfloat16* __restrict__ dst, int n) {
    int i = blockIdx.x * blockDim.x + threadIdx.x;
    if (i < n) dst[i] = __float2bfloat16(src[i]);
}

// ---------------------------------------------------------------------------
// Embedding gather (f32 -> bf16) + delta = clip(softplus(x . delta_proj))
// ---------------------------------------------------------------------------
__global__ __launch_bounds__(256) void embed_kernel(
    const int* __restrict__ tokens, const float* __restrict__ emb,
    const float* __restrict__ dproj, __hip_bfloat16* __restrict__ xb,
    float* __restrict__ delta) {
    int wave = threadIdx.x >> 5;
    int lane = threadIdx.x & 31;
    int row  = blockIdx.x * 8 + wave;
    int tok  = tokens[row];
    const float* e = emb + (size_t)tok * DD;
    __hip_bfloat16* xr = xb + (size_t)row * DD;
    float s = 0.f;
#pragma unroll
    for (int i = 0; i < DD / 32; ++i) {
        int d = lane + 32 * i;
        float v = e[d];
        xr[d] = __float2bfloat16(v);
        s += v * dproj[d];
    }
#pragma unroll
    for (int off = 16; off > 0; off >>= 1) s += __shfl_xor(s, off, 32);
    if (lane == 0) {
        float sp = (s > 20.f) ? s : __logf(1.f + __expf(s));   // softplus
        sp = fminf(fmaxf(sp, 0.001f), 0.1f);                   // clip
        delta[row] = sp;
    }
}

// ---------------------------------------------------------------------------
// LDS-staged bf16 WMMA GEMM:  C[M,N] = A[M,K] * W[N,K]^T  (+ epilogue)
//   MODE 0: silu -> bf16 (u);  MODE 1: -> bf16 (y);  MODE 2: +bias -> f32
// Block tile 128(M) x 128(N); 8 waves as 4(M) x 2(N), each wave 32x64 (8 acc).
// Per 32-K step: stage A(128x32) and B(128x32) slices into LDS with
// GLOBAL_LOAD_ASYNC_TO_LDS_B128 (double buffered, one barrier per step),
// fragments read back via ds_load_b128 with a conflict-free 80B pitch.
// grid = (M/128, N/128), block = 256.
// ---------------------------------------------------------------------------
template <int MODE, int K, int N>
__global__ __launch_bounds__(256) void gemm_kernel(
    const __hip_bfloat16* __restrict__ A,   // [M,K] row-major
    const __hip_bfloat16* __restrict__ W,   // [N,K] row-major (row = out col)
    float* __restrict__ Cf,                 // MODE 2 output
    __hip_bfloat16* __restrict__ Cb,        // MODE 0/1 output
    const float* __restrict__ bias) {       // MODE 2 bias
    __shared__ __hip_bfloat16 smem[2][2][128 * LDS_PITCH];  // [buf][A|B] 40KB

    const int tid  = threadIdx.x;
    const int wave = tid >> 5;
    const int lane = tid & 31;
    const int wm   = wave >> 1;   // 0..3 (M)
    const int wn   = wave & 1;    // 0..1 (N)
    const int r    = lane & 15;
    const int hi   = lane >> 4;

    // Staging: thread -> (row/col, 16B quarter of the 64B 32-K slice)
    const int scol = tid >> 2;    // 0..63 (second half handled at +64)
    const int sq   = tid & 3;
    const __hip_bfloat16* agsrc = A + (size_t)(blockIdx.x * 128 + scol) * K + sq * 8;
    const __hip_bfloat16* bgsrc = W + (size_t)(blockIdx.y * 128 + scol) * K + sq * 8;
    const int sdst = scol * LDS_PITCH + sq * 8;

    auto stage = [&](int buf, int k) {
        const __hip_bfloat16* ga = agsrc + k;
        const __hip_bfloat16* gb = bgsrc + k;
        __hip_bfloat16* la = &smem[buf][0][sdst];
        __hip_bfloat16* lb = &smem[buf][1][sdst];
        if (k + 96 <= K) {  // prefetch two steps ahead into near caches
            __builtin_prefetch((const void*)(ga + 64), 0, 3);
            __builtin_prefetch((const void*)(gb + 64), 0, 3);
        }
#if USE_ASYNC_LDS
        async_cp16(ga, la);
        async_cp16(ga + (size_t)64 * K, la + 64 * LDS_PITCH);
        async_cp16(gb, lb);
        async_cp16(gb + (size_t)64 * K, lb + 64 * LDS_PITCH);
#else
        *(v8bf*)la = *(const v8bf*)ga;
        *(v8bf*)(la + 64 * LDS_PITCH) = *(const v8bf*)(ga + (size_t)64 * K);
        *(v8bf*)lb = *(const v8bf*)gb;
        *(v8bf*)(lb + 64 * LDS_PITCH) = *(const v8bf*)(gb + (size_t)64 * K);
#endif
    };
    auto stage_wait = [&]() {
#if USE_ASYNC_LDS
#if __has_builtin(__builtin_amdgcn_s_wait_asynccnt)
        __builtin_amdgcn_s_wait_asynccnt(0);
#else
        asm volatile("s_wait_asynccnt 0" ::: "memory");
#endif
#endif
    };

    v8f acc[2][4] = {};

    stage(0, 0);
    stage_wait();
    __syncthreads();

    for (int k = 0; k < K; k += 32) {
        const int buf = (k >> 5) & 1;
        if (k + 32 < K) stage(buf ^ 1, k + 32);

        // A fragment (16x32): row (wm*32 + i*16 + r); lanes 0-15 hold elems
        // 0..7 & 16..23 (hi=0), lanes 16-31 hold 8..15 & 24..31 (hi=1).
        Frag fa[2];
#pragma unroll
        for (int i = 0; i < 2; ++i) {
            const __hip_bfloat16* p =
                &smem[buf][0][(wm * 32 + i * 16 + r) * LDS_PITCH + hi * 8];
            fa[i].h[0] = *(const v8bf*)(p);
            fa[i].h[1] = *(const v8bf*)(p + 16);
        }
        // B fragment (32x16): col (wn*64 + j*16 + r); lanes 0-15 hold elems
        // 0..7 & 16..23 (hi=0), lanes 16-31 hold 8..15 & 24..31 (hi=1).
        Frag fb[4];
#pragma unroll
        for (int j = 0; j < 4; ++j) {
            const __hip_bfloat16* p =
                &smem[buf][1][(wn * 64 + j * 16 + r) * LDS_PITCH + hi * 8];
            fb[j].h[0] = *(const v8bf*)(p);
            fb[j].h[1] = *(const v8bf*)(p + 16);
        }
#pragma unroll
        for (int i = 0; i < 2; ++i)
#pragma unroll
            for (int j = 0; j < 4; ++j)
                acc[i][j] = __builtin_amdgcn_wmma_f32_16x16x32_bf16(
                    false, fa[i].f, false, fb[j].f, (short)0, acc[i][j],
                    false, false);

        if (k + 32 < K) {
            stage_wait();
            __syncthreads();
        }
    }

    // C layout: VGPR rr, lanes 0-15 -> M = rr, lanes 16-31 -> M = rr + 8;
    // N = lane&15 within each 16-wide tile.
    const int m0 = blockIdx.x * 128 + wm * 32;
    const int n0 = blockIdx.y * 128 + wn * 64;
#pragma unroll
    for (int i = 0; i < 2; ++i) {
        const int mb = m0 + i * 16 + hi * 8;
#pragma unroll
        for (int j = 0; j < 4; ++j) {
            const int nn = n0 + j * 16 + r;
#pragma unroll
            for (int rr = 0; rr < 8; ++rr) {
                float v = acc[i][j][rr];
                const size_t idx = (size_t)(mb + rr) * N + nn;
                if (MODE == 0) {
                    float s = v / (1.f + __expf(-v));        // silu
                    Cb[idx] = __float2bfloat16(s);
                } else if (MODE == 1) {
                    Cb[idx] = __float2bfloat16(v);
                } else {
                    Cf[idx] = v + bias[nn];
                }
            }
        }
    }
}

// ---------------------------------------------------------------------------
// Chunked parallel scan of h_t = a_t h_{t-1} + (a_t-1)/denom * B_mat * u_t
// ---------------------------------------------------------------------------
__global__ __launch_bounds__(256) void scan_partial(
    const __hip_bfloat16* __restrict__ U, const float* __restrict__ delta,
    const float* __restrict__ A_log, const float* __restrict__ B_mat,
    float* __restrict__ Pk, float* __restrict__ Sk) {
    int idx = blockIdx.x * blockDim.x + threadIdx.x;  // b*CH*NN + c*NN + n
    int n = idx & (NN - 1);
    int c = (idx >> 10) & (CH - 1);
    int b = idx >> 14;
    float Abar  = __expf(A_log[n]);
    float scale = B_mat[n] / fmaxf(fabsf(Abar), 1e-8f);
    const float* dl = delta + (size_t)b * TT + c * TC;
    size_t base = ((size_t)b * TT + (size_t)c * TC) * NN + n;
    float P = 1.f, S = 0.f;
#pragma unroll 4
    for (int t = 0; t < TC; ++t) {
        float a = __expf(dl[t] * Abar);
        float u = __bfloat162float(U[base + (size_t)t * NN]);
        S = fmaf(a, S, (a - 1.f) * scale * u);
        P *= a;
    }
    Pk[idx] = P;
    Sk[idx] = S;
}

__global__ __launch_bounds__(256) void scan_carry(
    const float* __restrict__ Pk, const float* __restrict__ Sk,
    float* __restrict__ H0) {
    int idx = blockIdx.x * blockDim.x + threadIdx.x;  // 0 .. BB*NN-1
    int n = idx & (NN - 1);
    int b = idx >> 10;
    float h = 0.f;
#pragma unroll
    for (int c = 0; c < CH; ++c) {
        int j = (b * CH + c) * NN + n;
        H0[j] = h;
        h = fmaf(Pk[j], h, Sk[j]);
    }
}

__global__ __launch_bounds__(256) void scan_final(
    const __hip_bfloat16* __restrict__ U, const float* __restrict__ delta,
    const float* __restrict__ A_log, const float* __restrict__ B_mat,
    const float* __restrict__ H0, __hip_bfloat16* __restrict__ HS) {
    int idx = blockIdx.x * blockDim.x + threadIdx.x;
    int n = idx & (NN - 1);
    int c = (idx >> 10) & (CH - 1);
    int b = idx >> 14;
    float Abar  = __expf(A_log[n]);
    float scale = B_mat[n] / fmaxf(fabsf(Abar), 1e-8f);
    const float* dl = delta + (size_t)b * TT + c * TC;
    size_t base = ((size_t)b * TT + (size_t)c * TC) * NN + n;
    float h = H0[idx];
#pragma unroll 4
    for (int t = 0; t < TC; ++t) {
        float a = __expf(dl[t] * Abar);
        float u = __bfloat162float(U[base + (size_t)t * NN]);
        h = fmaf(a, h, (a - 1.f) * scale * u);
        HS[base + (size_t)t * NN] = __float2bfloat16(h);
    }
}

// ---------------------------------------------------------------------------
// Host-side launch
// ---------------------------------------------------------------------------
extern "C" void kernel_launch(void* const* d_in, const int* in_sizes, int n_in,
                              void* d_out, int out_size, void* d_ws, size_t ws_size,
                              hipStream_t stream) {
    const int*   tokens    = (const int*)d_in[0];
    const float* embedding = (const float*)d_in[1];
    const float* W_in      = (const float*)d_in[2];
    const float* W_out     = (const float*)d_in[3];
    const float* A_log     = (const float*)d_in[4];
    const float* B_mat     = (const float*)d_in[5];
    const float* dproj     = (const float*)d_in[6];
    const float* head_W    = (const float*)d_in[7];
    const float* head_b    = (const float*)d_in[8];
    float* out = (float*)d_out;

    // Workspace partition (256B aligned)
    char* ws = (char*)d_ws;
    size_t off = 0;
    auto alloc = [&](size_t bytes) -> void* {
        void* p = ws + off;
        off = (off + bytes + 255) & ~(size_t)255;
        return p;
    };
    __hip_bfloat16* XB    = (__hip_bfloat16*)alloc((size_t)MM * DD * 2);
    float*          DELTA = (float*)alloc((size_t)MM * 4);
    __hip_bfloat16* WINB  = (__hip_bfloat16*)alloc((size_t)NN * DD * 2);
    __hip_bfloat16* WOUTB = (__hip_bfloat16*)alloc((size_t)DD * NN * 2);
    __hip_bfloat16* HEADB = (__hip_bfloat16*)alloc((size_t)VV * DD * 2);
    __hip_bfloat16* U     = (__hip_bfloat16*)alloc((size_t)MM * NN * 2);
    __hip_bfloat16* HS    = (__hip_bfloat16*)alloc((size_t)MM * NN * 2);
    __hip_bfloat16* Y     = (__hip_bfloat16*)alloc((size_t)MM * DD * 2);
    float*          Pk    = (float*)alloc((size_t)BB * CH * NN * 4);
    float*          Sk    = (float*)alloc((size_t)BB * CH * NN * 4);
    float*          H0    = (float*)alloc((size_t)BB * CH * NN * 4);
    (void)ws_size; (void)in_sizes; (void)n_in; (void)out_size;

    // 1) weight conversions to bf16
    cvt_f32_bf16<<<(NN * DD + 255) / 256, 256, 0, stream>>>(W_in, WINB, NN * DD);
    cvt_f32_bf16<<<(DD * NN + 255) / 256, 256, 0, stream>>>(W_out, WOUTB, DD * NN);
    cvt_f32_bf16<<<(VV * DD + 255) / 256, 256, 0, stream>>>(head_W, HEADB, VV * DD);

    // 2) embedding gather + delta
    embed_kernel<<<MM / 8, 256, 0, stream>>>(tokens, embedding, dproj, XB, DELTA);

    // 3) u = silu(x W_in^T)   [MM,DD] x [NN,DD]^T
    gemm_kernel<0, DD, NN><<<dim3(MM / 128, NN / 128), 256, 0, stream>>>(
        XB, WINB, nullptr, U, nullptr);

    // 4) chunked parallel scan over T
    scan_partial<<<(BB * CH * NN) / 256, 256, 0, stream>>>(U, DELTA, A_log, B_mat, Pk, Sk);
    scan_carry<<<(BB * NN) / 256, 256, 0, stream>>>(Pk, Sk, H0);
    scan_final<<<(BB * CH * NN) / 256, 256, 0, stream>>>(U, DELTA, A_log, B_mat, H0, HS);

    // 5) y = hs W_out^T       [MM,NN] x [DD,NN]^T
    gemm_kernel<1, NN, DD><<<dim3(MM / 128, DD / 128), 256, 0, stream>>>(
        HS, WOUTB, nullptr, Y, nullptr);

    // 6) logits = y head_W^T + bias   [MM,DD] x [VV,DD]^T
    gemm_kernel<2, DD, VV><<<dim3(MM / 128, VV / 128), 256, 0, stream>>>(
        Y, HEADB, out, nullptr, head_b);
}
